// PolicyNetAtt_17248588660954
// MI455X (gfx1250) — matrix operand order repaired and checked by reading, not compiled, not run on hardware
//
#include <hip/hip_runtime.h>
#include <math.h>

typedef __attribute__((ext_vector_type(16))) _Float16 v16h;
typedef __attribute__((ext_vector_type(8)))  _Float16 h8;
typedef __attribute__((ext_vector_type(2)))  _Float16 h2;
typedef __attribute__((ext_vector_type(8)))  float    v8f;

#define WPB 8            // waves per block (256 threads, wave32)
#define LE1_STRIDE 80    // halves per LDS row (64 data + 16 pad) -> 160B row, 16B aligned

// ---------------------------------------------------------------------------
// Kernel 1: per-landmark MLP (WMMA) + online softmax partials per block.
// Each wave owns tiles of 16 landmarks (grid-stride).
// ---------------------------------------------------------------------------
__global__ __launch_bounds__(256) void policy_lm_kernel(
    const float* __restrict__ obs,
    const float* __restrict__ ap_w1, const float* __restrict__ ap_b1,
    const float* __restrict__ ap_w2, const float* __restrict__ ap_b2,
    const float* __restrict__ lm_w1, const float* __restrict__ lm_b1,
    const float* __restrict__ lm_w2, const float* __restrict__ lm_b2,
    float* __restrict__ partials, int nl, int ntiles)
{
  __shared__ __align__(16) _Float16 s_le1[WPB * 16 * LE1_STRIDE];
  __shared__ float s_h1[32];
  __shared__ float s_agent[32];
  __shared__ float s_red[WPB][34];

  const int tid  = threadIdx.x;
  const int lane = tid & 31;
  const int wv   = tid >> 5;
  const int col  = lane & 15;   // 0..15
  const int half = lane >> 4;   // 0 or 1

  // ---- tiny agent MLP: [3]->32->32, computed once per block by wave 0 ----
  if (wv == 0) {
    float o0 = obs[0], o1 = obs[1], o2 = obs[2];
    float h = ap_w1[lane*3+0]*o0 + ap_w1[lane*3+1]*o1 + ap_w1[lane*3+2]*o2 + ap_b1[lane];
    s_h1[lane] = fmaxf(h, 0.0f);
  }
  __syncthreads();
  if (wv == 0) {
    float a = ap_b2[lane];
#pragma unroll
    for (int j = 0; j < 32; ++j) a += ap_w2[lane*32+j] * s_h1[j];
    s_agent[lane] = fmaxf(a, 0.0f);
  }
  __syncthreads();
  const float ag0 = s_agent[col];
  const float ag1 = s_agent[16 + col];

  // ---- per-wave constant WMMA fragments ----
  // Layer-1 B (lm_w1^T as 32x64 f16, K padded 4->32), 4 N-tiles.
  // B layout: lanes 0-15 hold K=0..15 (elem e = K e), lanes 16-31 hold K=16..31 (all zero here).
  v16h B1[4];
#pragma unroll
  for (int t = 0; t < 4; ++t) {
    v16h b = {};
    if (half == 0) {
      const float* w = lm_w1 + (t*16 + col) * 4;
      b[0] = (_Float16)w[0]; b[1] = (_Float16)w[1];
      b[2] = (_Float16)w[2]; b[3] = (_Float16)w[3];
    }
    B1[t] = b;
  }
  float bias1[4];
#pragma unroll
  for (int t = 0; t < 4; ++t) bias1[t] = lm_b1[t*16 + col];

  // Layer-2 B (lm_w2^T as 64x32 f16): 2 K-chunks x 2 N-tiles.
  // lane: N = nt*16+col, elem e -> K = kt*32 + half*16 + e.
  v16h B2[2][2];
#pragma unroll
  for (int kt = 0; kt < 2; ++kt)
#pragma unroll
    for (int nt = 0; nt < 2; ++nt) {
      const float* w = lm_w2 + (nt*16 + col) * 64 + kt*32 + half*16;
      v16h b;
#pragma unroll
      for (int e = 0; e < 16; ++e) b[e] = (_Float16)w[e];
      B2[kt][nt] = b;
    }
  float bias2[2];
  bias2[0] = lm_b2[col];
  bias2[1] = lm_b2[16 + col];

  const float* info_base = obs + 3;
  const float* est_base  = obs + 3 + 2*nl;
  const float* mask_base = obs + 3 + 4*nl;

  _Float16* lds = &s_le1[wv * 16 * LE1_STRIDE];

  const int gw = blockIdx.x * WPB + wv;
  const int nw = gridDim.x * WPB;

  // online-softmax running state (per half-wave, columns per lane)
  float runmax = -3.0e38f, runsum = 0.0f;
  float vacc0 = 0.0f, vacc1 = 0.0f;

  for (int tile = gw; tile < ntiles; tile += nw) {
    const int base = tile * 16;
    const int l    = base + col;
    const bool valid = (l < nl);
    const int lc   = valid ? l : (nl - 1);

    // lanes 0-15 load est pair + mask, lanes 16-31 load info pair (coalesced b32)
    float px, py, maskv = 0.0f;
    if (half == 0) {
      px = est_base[2*lc + 0];
      py = est_base[2*lc + 1];
      maskv = valid ? mask_base[lc] : 0.0f;
    } else {
      px = info_base[2*lc + 0];
      py = info_base[2*lc + 1];
    }
    {
      int pf = tile + nw;
      if (pf < ntiles) __builtin_prefetch(est_base + 2*(pf*16 + col), 0, 0);
    }

    // build layer-1 A (16x32 f16, K=0..3 live): lane<16 holds row M=col
    union { h2 h; int i; } u;
    u.h[0] = (_Float16)px; u.h[1] = (_Float16)py;
    union { int i; h2 h; } o;
    o.i = __shfl_xor(u.i, 16, 32);
    v16h A1 = {};
    if (half == 0) {
      A1[0] = u.h[0]; A1[1] = u.h[1];   // K0,K1 = est
      A1[2] = o.h[0]; A1[3] = o.h[1];   // K2,K3 = info
    }

    // ---- layer 1: le1 = relu(A1 @ W1^T + b1), 16x64, via 4 WMMAs ----
#pragma unroll
    for (int t = 0; t < 4; ++t) {
      v8f c = { bias1[t], bias1[t], bias1[t], bias1[t],
                bias1[t], bias1[t], bias1[t], bias1[t] };
      c = __builtin_amdgcn_wmma_f32_16x16x32_f16(false, A1, false, B1[t],
                                                 (short)0, c, false, false);
      // C layout: vgpr r -> row M=r+8*half, col N=t*16+col ; store f16 to LDS
#pragma unroll
      for (int r = 0; r < 8; ++r) {
        lds[(r + 8*half) * LE1_STRIDE + t*16 + col] = (_Float16)fmaxf(c[r], 0.0f);
      }
    }

    // ---- transpose read: build layer-2 A fragments (row M=col, K along lane) ----
    const _Float16* lr = lds + col * LE1_STRIDE;
    const int koff = half * 16;
    h8 p0 = *(const h8*)(lr + koff);            // K = koff..koff+7
    h8 p1 = *(const h8*)(lr + koff + 8);        // K = koff+8..+15
    h8 p2 = *(const h8*)(lr + 32 + koff);       // K = 32+koff..
    h8 p3 = *(const h8*)(lr + 40 + koff);       // K = 40+koff..
    v16h A2a, A2b;
#pragma unroll
    for (int e = 0; e < 8; ++e) {
      A2a[e] = p0[e]; A2a[8+e] = p1[e];
      A2b[e] = p2[e]; A2b[8+e] = p3[e];
    }

    // ---- layer 2: le2 = relu(le1 @ W2^T + b2), 16x32, via 4 WMMAs ----
    v8f c0 = { bias2[0], bias2[0], bias2[0], bias2[0],
               bias2[0], bias2[0], bias2[0], bias2[0] };
    v8f c1 = { bias2[1], bias2[1], bias2[1], bias2[1],
               bias2[1], bias2[1], bias2[1], bias2[1] };
    c0 = __builtin_amdgcn_wmma_f32_16x16x32_f16(false, A2a, false, B2[0][0], (short)0, c0, false, false);
    c0 = __builtin_amdgcn_wmma_f32_16x16x32_f16(false, A2b, false, B2[1][0], (short)0, c0, false, false);
    c1 = __builtin_amdgcn_wmma_f32_16x16x32_f16(false, A2a, false, B2[0][1], (short)0, c1, false, false);
    c1 = __builtin_amdgcn_wmma_f32_16x16x32_f16(false, A2b, false, B2[1][1], (short)0, c1, false, false);
#pragma unroll
    for (int r = 0; r < 8; ++r) { c0[r] = fmaxf(c0[r], 0.0f); c1[r] = fmaxf(c1[r], 0.0f); }

    // ---- attention logits: dot(agent, le2[m]) over 32 cols split across 16 lanes ----
    float p[8];
#pragma unroll
    for (int r = 0; r < 8; ++r) p[r] = ag0 * c0[r] + ag1 * c1[r];
#pragma unroll
    for (int s = 1; s < 16; s <<= 1)
#pragma unroll
      for (int r = 0; r < 8; ++r) p[r] += __shfl_xor(p[r], s, 32);

    // mask + scale; p[r] is logit of landmark base + 8*half + r (replicated in half)
    float tmax = -3.0e38f;
#pragma unroll
    for (int r = 0; r < 8; ++r) {
      float a = p[r] * 0.25f;
      float mv = __shfl(maskv, 8*half + r, 32);
      a = (mv == 0.0f) ? -1.0e10f : a;
      a = ((base + 8*half + r) < nl) ? a : -3.0e38f;
      p[r] = a;
      tmax = fmaxf(tmax, a);
    }

    // online softmax update (per half-wave)
    float nmax = fmaxf(runmax, tmax);
    float sc = __expf(runmax - nmax);
    float ts = 0.0f, tv0 = 0.0f, tv1 = 0.0f;
#pragma unroll
    for (int r = 0; r < 8; ++r) {
      float w = __expf(p[r] - nmax);
      ts += w; tv0 += w * c0[r]; tv1 += w * c1[r];
    }
    runsum = runsum * sc + ts;
    vacc0  = vacc0  * sc + tv0;
    vacc1  = vacc1  * sc + tv1;
    runmax = nmax;
  }

  // ---- combine halves of the wave (exp-rescale merge) ----
  float om  = __shfl_xor(runmax, 16, 32);
  float M2  = fmaxf(runmax, om);
  float f   = __expf(runmax - M2);
  float ss  = runsum * f;
  float S2  = ss + __shfl_xor(ss, 16, 32);
  float v0s = vacc0 * f, v1s = vacc1 * f;
  float V0  = v0s + __shfl_xor(v0s, 16, 32);
  float V1  = v1s + __shfl_xor(v1s, 16, 32);

  if (lane < 16) { s_red[wv][2 + col] = V0; s_red[wv][18 + col] = V1; }
  if (lane == 0) { s_red[wv][0] = M2; s_red[wv][1] = S2; }
  __syncthreads();

  // ---- combine waves of the block, write block partial (36-float slot) ----
  if (wv == 0) {
    float bm = -3.0e38f;
#pragma unroll
    for (int w = 0; w < WPB; ++w) bm = fmaxf(bm, s_red[w][0]);
    float bs = 0.0f, bv = 0.0f;
#pragma unroll
    for (int w = 0; w < WPB; ++w) {
      float g = __expf(s_red[w][0] - bm);
      bs += g * s_red[w][1];
      bv += g * s_red[w][2 + lane];
    }
    float* outp = partials + (size_t)blockIdx.x * 36;
    if (lane == 0) { outp[0] = bm; outp[1] = bs; }
    outp[2 + lane] = bv;
  }
}

// ---------------------------------------------------------------------------
// Kernel 2: merge block partials, run the tiny head MLP, write 2 outputs.
// ---------------------------------------------------------------------------
__global__ __launch_bounds__(64) void policy_head_kernel(
    const float* __restrict__ partials, int nb,
    const float* __restrict__ obs,
    const float* __restrict__ ap_w1, const float* __restrict__ ap_b1,
    const float* __restrict__ ap_w2, const float* __restrict__ ap_b2,
    const float* __restrict__ info_w, const float* __restrict__ info_b,
    const float* __restrict__ act_w1, const float* __restrict__ act_b1,
    const float* __restrict__ act_w2, const float* __restrict__ act_b2,
    float* __restrict__ out)
{
  __shared__ float sh[32];   // agent hidden
  __shared__ float sx[64];   // [agent, lm_att] then reused for tanh layer
  __shared__ float se[64];   // info_emb

  const int tid = threadIdx.x;

  if (tid < 32) {
    // global softmax merge across blocks
    float m = -3.0e38f;
    for (int b = 0; b < nb; ++b) m = fmaxf(m, partials[(size_t)b * 36]);
    float S = 0.0f, V = 0.0f;
    for (int b = 0; b < nb; ++b) {
      const float* p = partials + (size_t)b * 36;
      float g = __expf(p[0] - m);
      S += g * p[1];
      V += g * p[2 + tid];
    }
    sx[32 + tid] = fmaxf(V / S, 0.0f);   // lm_att

    // agent layer 1
    float h = ap_w1[tid*3+0]*obs[0] + ap_w1[tid*3+1]*obs[1] + ap_w1[tid*3+2]*obs[2] + ap_b1[tid];
    sh[tid] = fmaxf(h, 0.0f);
  }
  __syncthreads();
  if (tid < 32) {
    float a = ap_b2[tid];
#pragma unroll
    for (int j = 0; j < 32; ++j) a += ap_w2[tid*32+j] * sh[j];
    sx[tid] = fmaxf(a, 0.0f);            // agent
  }
  __syncthreads();

  {
    float a = info_b[tid];
#pragma unroll
    for (int j = 0; j < 64; ++j) a += info_w[tid*64+j] * sx[j];
    se[tid] = fmaxf(a, 0.0f);            // info_emb
  }
  __syncthreads();
  {
    float a = act_b1[tid];
#pragma unroll
    for (int j = 0; j < 64; ++j) a += act_w1[tid*64+j] * se[j];
    sx[tid] = tanhf(a);
  }
  __syncthreads();
  if (tid < 2) {
    float a = act_b2[tid];
#pragma unroll
    for (int j = 0; j < 64; ++j) a += act_w2[tid*64+j] * sx[j];
    a = tanhf(a);
    out[tid] = (tid == 0) ? (1.0f + a) * 2.0f : a * 1.0471975511965976f; // pi/3
  }
}

// ---------------------------------------------------------------------------
extern "C" void kernel_launch(void* const* d_in, const int* in_sizes, int n_in,
                              void* d_out, int out_size, void* d_ws, size_t ws_size,
                              hipStream_t stream) {
  const float* obs    = (const float*)d_in[0];
  const float* ap_w1  = (const float*)d_in[1];
  const float* ap_b1  = (const float*)d_in[2];
  const float* ap_w2  = (const float*)d_in[3];
  const float* ap_b2  = (const float*)d_in[4];
  const float* lm_w1  = (const float*)d_in[5];
  const float* lm_b1  = (const float*)d_in[6];
  const float* lm_w2  = (const float*)d_in[7];
  const float* lm_b2  = (const float*)d_in[8];
  const float* info_w = (const float*)d_in[9];
  const float* info_b = (const float*)d_in[10];
  const float* act_w1 = (const float*)d_in[11];
  const float* act_b1 = (const float*)d_in[12];
  const float* act_w2 = (const float*)d_in[13];
  const float* act_b2 = (const float*)d_in[14];
  float* outp = (float*)d_out;

  const int D = in_sizes[0];
  const int nl = (D - 3) / 5;
  const int ntiles = (nl + 15) / 16;

  int nb = 512;                                     // blocks (x8 waves each)
  int need = (ntiles + WPB - 1) / WPB;              // don't launch idle blocks
  if (nb > need) nb = need;
  int cap = (int)(ws_size / (36 * sizeof(float)));  // workspace bound
  if (nb > cap) nb = cap;
  if (nb < 1) nb = 1;

  float* partials = (float*)d_ws;

  policy_lm_kernel<<<nb, 256, 0, stream>>>(
      obs, ap_w1, ap_b1, ap_w2, ap_b2,
      lm_w1, lm_b1, lm_w2, lm_b2,
      partials, nl, ntiles);

  policy_head_kernel<<<1, 64, 0, stream>>>(
      partials, nb, obs,
      ap_w1, ap_b1, ap_w2, ap_b2,
      info_w, info_b, act_w1, act_b1, act_w2, act_b2,
      outp);
}
